// Autoformer_19542101197528
// MI455X (gfx1250) — compile-verified, matrix operand order checked by reading, not run on hardware
//
#include <hip/hip_runtime.h>
#include <hip/hip_bf16.h>
#include <math.h>

// ---------------------------------------------------------------------------
// Autoformer decoder for MI455X (gfx1250, wave32).
// - FFT auto-correlation replaced by exact circular correlation: the mean over
//   heads*channels collapses it to mv[b,l] = (1/64) sum_t <q[(t+l)%L], k[t]>,
//   i.e. wrapped-diagonal sums of the Gram matrix G = Q K^T -> WMMA.
// - Q,K (221KB/batch) staged whole in CDNA5's 320KB LDS via
//   global_load_async_to_lds_b128 + s_wait_asynccnt.
// - All dense GEMMs on the matrix pipe: v_wmma_f32_16x16x32_f16 (f32 acc).
// Workspace: 8 * 512*432*64 floats + ~2MB small scratch (~455MB).
// ---------------------------------------------------------------------------

#define BNB  512            // batch*nodes
#define LD   432            // L_DEC
#define DM   64             // d_model
#define PRED 144
#define TOK  (BNB * LD)     // 221184 tokens (divisible by 16)
#define NTOPK 4             // int(0.8 * ln(432)) == 4
#define CPAD 68             // LDS row pitch (floats): 16B-aligned, non-pow2 banks

typedef __attribute__((ext_vector_type(16))) _Float16 v16h;
typedef __attribute__((ext_vector_type(8)))  float    v8f;

// ---------------------------------------------------------------------------
// WMMA linear layer:  Y[t,n] = epilogue( sum_k X[t,k] * W[n,k] )
// One 16(M)x16(N) tile per wave. K in {64,128}, N in {64,128}.
// A-fragment layout per CDNA5 ISA (16-bit A 16x32):
//   lane%16 = M, regs 0..3 hold K=2v,2v+1 (+8 if lane>=16), regs 4..7 K=16+...
// B mirrored with lane%16 = N column (row of W, since Y = X*W^T).
// C/D: reg r -> row = 16*tm + r + 8*(lane>=16), col = 16*tn + (lane&15).
// ---------------------------------------------------------------------------
template<int K, int N, bool BIAS, bool GELU_ACT, bool RES>
__global__ void __launch_bounds__(256)
linear_wmma(const float* __restrict__ X, const float* __restrict__ W,
            const float* __restrict__ bias, const float* __restrict__ Rsd,
            float* __restrict__ Y, int T)
{
    const int lane = threadIdx.x & 31;
    const int wave = threadIdx.x >> 5;
    const int tiles_n = N / 16;
    const long tiles_total = (long)(T / 16) * tiles_n;
    const long tile = (long)blockIdx.x * (blockDim.x >> 5) + wave;
    if (tile >= tiles_total) return;              // wave-uniform exit

    const int  tn = (int)(tile % tiles_n);
    const long tm = tile / tiles_n;
    const int  mrow = lane & 15;
    const int  ksel = (lane >> 4) * 8;            // K offset +8 for upper half-wave

    const float* xrow = X + (tm * 16 + mrow) * (long)K;
    const float* wrow = W + (tn * 16 + mrow) * (long)K;
    __builtin_prefetch(xrow, 0, 1);               // global_prefetch_b8 (streaming A)

    v8f acc = {};
#pragma unroll
    for (int k0 = 0; k0 < K; k0 += 32) {
        v16h a, b;
#pragma unroll
        for (int v = 0; v < 8; ++v) {
            const int kk = k0 + ((v < 4) ? 2 * v : 16 + 2 * (v - 4)) + ksel;
            a[2 * v]     = (_Float16)xrow[kk];
            a[2 * v + 1] = (_Float16)xrow[kk + 1];
            b[2 * v]     = (_Float16)wrow[kk];
            b[2 * v + 1] = (_Float16)wrow[kk + 1];
        }
        acc = __builtin_amdgcn_wmma_f32_16x16x32_f16(
                  false, a, false, b, (short)0, acc, false, false);
    }

#pragma unroll
    for (int r = 0; r < 8; ++r) {
        const long row = tm * 16 + r + (lane >> 4) * 8;
        const int  col = tn * 16 + (lane & 15);
        float v = acc[r];
        if constexpr (BIAS)     v += bias[col];
        if constexpr (GELU_ACT) v = 0.5f * v * (1.0f + erff(v * 0.70710678118f));
        if constexpr (RES)      v += Rsd[row * N + col];
        Y[row * N + col] = v;
    }
}

// ---------------------------------------------------------------------------
// conv stems (direct; 21/192/144 MACs per output -> VALU is right)
// x[bn,ci,h,w] = day_seq[bn, h, w, ci]
// ---------------------------------------------------------------------------
__global__ void conv0_kernel(const float* __restrict__ ds, const float* __restrict__ w,
                             const float* __restrict__ bias, float* __restrict__ out)
{
    const long i = (long)blockIdx.x * 256 + threadIdx.x;   // BN*432*64
    if (i >= (long)TOK * DM) return;
    const int co = (int)(i & 63);
    const long bl = i >> 6;
    const int l = (int)(bl % LD), bn = (int)(bl / LD);
    const int ho = l / 144, wv = l % 144;                  // flat offset 0 (3*144=432)
    const float* xb = ds + (long)bn * (7 * 144 * 3);
    float s = bias[co];
    for (int ci = 0; ci < 3; ++ci)
        for (int kh = 0; kh < 7; ++kh) {
            const int h = ho + kh - 1;                     // pad_h = 1
            if (h < 0 || h > 6) continue;
            s += xb[(h * 144 + wv) * 3 + ci] * w[(co * 3 + ci) * 7 + kh];
        }
    out[i] = s;
}

__global__ void conv1_kernel(const float* __restrict__ ds, const float* __restrict__ w,
                             const float* __restrict__ bias, float* __restrict__ out)
{
    const long i = (long)blockIdx.x * 256 + threadIdx.x;
    if (i >= (long)TOK * DM) return;
    const int co = (int)(i & 63);
    const long bl = i >> 6;
    const int l = (int)(bl % LD), bn = (int)(bl / LD);
    const int f = l + 224;                                 // 8*82 - 432
    const int ho = f / 82, wo = f % 82;                    // W_out = 144-63+1
    const float* xb = ds + (long)bn * (7 * 144 * 3);
    float s = bias[co];
    for (int ci = 0; ci < 3; ++ci)
        for (int kh = 0; kh < 2; ++kh) {
            const int h = ho + kh - 1;                     // pad_h = 1
            if (h < 0 || h > 6) continue;
            const float* xr = xb + (h * 144 + wo) * 3 + ci;
            const float* wr = w + ((co * 3 + ci) * 2 + kh) * 32;
            for (int kw = 0; kw < 32; ++kw)                // dilation_w = 2
                s += xr[2 * kw * 3] * wr[kw];
        }
    out[i] = s;
}

__global__ void conv2_kernel(const float* __restrict__ ds, const float* __restrict__ w,
                             const float* __restrict__ bias, float* __restrict__ out)
{
    const long i = (long)blockIdx.x * 256 + threadIdx.x;
    if (i >= (long)TOK * DM) return;
    const int co = (int)(i & 63);
    const long bl = i >> 6;
    const int l = (int)(bl % LD), bn = (int)(bl / LD);
    const int f = l + 247;                                 // 7*97 - 432
    const int ho = f / 97, wo = f % 97;                    // W_out = 144-48+1
    const float* xb = ds + (long)bn * (7 * 144 * 3);
    float s = bias[co];
    for (int ci = 0; ci < 3; ++ci) {
        const float* xr = xb + (ho * 144 + wo) * 3 + ci;
        const float* wr = w + (co * 3 + ci) * 48;
        for (int kw = 0; kw < 48; ++kw)
            s += xr[kw * 3] * wr[kw];
    }
    out[i] = s;
}

// ---------------------------------------------------------------------------
// Circular correlation via WMMA Gram matrix + wrapped-diagonal reduction.
// One workgroup (32 waves) per batch b. Q,K rows async-copied into LDS
// (row pitch CPAD=68 floats). 27x27 16x16 WMMA tiles of G = Q K^T; element
// (row,col) accumulates into lag (row-col) mod 432 via ds_add_f32.
// ---------------------------------------------------------------------------
__global__ void __launch_bounds__(1024)
corr_wmma_kernel(const float* __restrict__ Q, const float* __restrict__ Kb,
                 float* __restrict__ MV)
{
    extern __shared__ float smem[];                // [LD*CPAD | LD*CPAD | LD]
    const int b    = blockIdx.x;
    const int tid  = threadIdx.x;
    const int lane = tid & 31;
    const int wave = tid >> 5;

    const unsigned QBYTES = (unsigned)(LD * CPAD * 4);
    float* smv = smem + 2 * LD * CPAD;

    // stage Q and K (16B chunks, padded rows) through the async-to-LDS path
    for (int idx = tid; idx < 2 * LD * 16; idx += 1024) {
        const int isK = idx >= LD * 16;
        const int rc  = isK ? idx - LD * 16 : idx;
        const int r   = rc >> 4;
        const int c   = rc & 15;
        const float* src = (isK ? Kb : Q) + (long)b * LD * DM + r * DM + c * 4;
        const unsigned dst = (isK ? QBYTES : 0u) + (unsigned)(r * CPAD + c * 4) * 4u;
        asm volatile("global_load_async_to_lds_b128 %0, %1, off"
                     :: "v"(dst), "v"(src) : "memory");
    }
    if (tid < LD) smv[tid] = 0.0f;
    asm volatile("s_wait_asynccnt 0" ::: "memory");
    __syncthreads();

    const float* sQ = smem;
    const float* sK = smem + LD * CPAD;
    const int mrow = lane & 15;
    const int ksel = (lane >> 4) * 8;

    for (int tile = wave; tile < 729; tile += 32) {        // wave-uniform trip
        const int tm = tile / 27, tn = tile % 27;
        const float* qr = sQ + (tm * 16 + mrow) * CPAD;
        const float* kr = sK + (tn * 16 + mrow) * CPAD;
        v8f acc = {};
#pragma unroll
        for (int k0 = 0; k0 < DM; k0 += 32) {
            v16h a, bb;
#pragma unroll
            for (int v = 0; v < 8; ++v) {
                const int kk = k0 + ((v < 4) ? 2 * v : 16 + 2 * (v - 4)) + ksel;
                a[2 * v]      = (_Float16)qr[kk];
                a[2 * v + 1]  = (_Float16)qr[kk + 1];
                bb[2 * v]     = (_Float16)kr[kk];
                bb[2 * v + 1] = (_Float16)kr[kk + 1];
            }
            acc = __builtin_amdgcn_wmma_f32_16x16x32_f16(
                      false, a, false, bb, (short)0, acc, false, false);
        }
#pragma unroll
        for (int r = 0; r < 8; ++r) {
            const int row = tm * 16 + r + (lane >> 4) * 8;
            const int col = tn * 16 + (lane & 15);
            int lag = row - col; if (lag < 0) lag += LD;
            atomicAdd(&smv[lag], acc[r]);                  // ds_add_f32
        }
    }
    __syncthreads();
    if (tid < LD) MV[b * LD + tid] = smv[tid] * (1.0f / 64.0f);
}

// top-4 of mean over batch of mv  (matches jax.lax.top_k ordering: descending)
__global__ void topk_kernel(const float* __restrict__ MV, int* __restrict__ IDX)
{
    __shared__ float m[LD];
    const int l = threadIdx.x;
    if (l < LD) {
        float s = 0.f;
        for (int b = 0; b < BNB; ++b) s += MV[b * LD + l];
        m[l] = s * (1.0f / BNB);
    }
    __syncthreads();
    if (threadIdx.x == 0) {
        for (int k = 0; k < NTOPK; ++k) {
            int bi = 0; float bv = -1e30f;
            for (int j = 0; j < LD; ++j) { const float v = m[j]; if (v > bv) { bv = v; bi = j; } }
            IDX[k] = bi; m[bi] = -1e30f;
        }
    }
}

__global__ void softmax_kernel(const float* __restrict__ MV, const int* __restrict__ IDX,
                               float* __restrict__ CW)
{
    const int b = blockIdx.x * blockDim.x + threadIdx.x;
    if (b >= BNB) return;
    float w[NTOPK], mx = -1e30f;
    for (int k = 0; k < NTOPK; ++k) { w[k] = MV[b * LD + IDX[k]]; mx = fmaxf(mx, w[k]); }
    float s = 0.f;
    for (int k = 0; k < NTOPK; ++k) { w[k] = __expf(w[k] - mx); s += w[k]; }
    const float inv = 1.0f / s;
    for (int k = 0; k < NTOPK; ++k) CW[b * NTOPK + k] = w[k] * inv;
}

// agg[b,l,:] = sum_k cw[b,k] * v[b, (l+idx[k])%L, :]
__global__ void agg_kernel(const float* __restrict__ V, const int* __restrict__ IDX,
                           const float* __restrict__ CW, float* __restrict__ OUT)
{
    const long i = (long)blockIdx.x * 256 + threadIdx.x;   // TOK * 16 (float4 over d)
    if (i >= (long)TOK * 16) return;
    const int d4 = (int)(i & 15);
    const long bl = i >> 4;
    const int l = (int)(bl % LD), b = (int)(bl / LD);
    float ax = 0.f, ay = 0.f, az = 0.f, aw = 0.f;
    for (int k = 0; k < NTOPK; ++k) {
        int src = l + IDX[k]; if (src >= LD) src -= LD;
        const float4 v = ((const float4*)(V + ((long)b * LD + src) * DM))[d4];
        const float c = CW[b * NTOPK + k];
        ax += c * v.x; ay += c * v.y; az += c * v.z; aw += c * v.w;
    }
    float4 r; r.x = ax; r.y = ay; r.z = az; r.w = aw;
    ((float4*)(OUT + bl * DM))[d4] = r;
}

// series_decomp: season = x - movavg25(x); TACC += movavg (edge-replicate pad)
__global__ void decomp_kernel(const float* __restrict__ X, float* __restrict__ SEASON,
                              float* __restrict__ TACC)
{
    const long i = (long)blockIdx.x * 256 + threadIdx.x;   // TOK * 16
    if (i >= (long)TOK * 16) return;
    const int d4 = (int)(i & 15);
    const long bl = i >> 4;
    const int l = (int)(bl % LD), b = (int)(bl / LD);
    const float4* xb = (const float4*)(X + (long)b * LD * DM);
    float sx = 0.f, sy = 0.f, sz = 0.f, sw = 0.f;
    for (int j = 0; j < 25; ++j) {
        int ll = l + j - 12;
        ll = ll < 0 ? 0 : (ll > LD - 1 ? LD - 1 : ll);
        const float4 v = xb[(long)ll * 16 + d4];
        sx += v.x; sy += v.y; sz += v.z; sw += v.w;
    }
    const float inv = 1.0f / 25.0f;
    sx *= inv; sy *= inv; sz *= inv; sw *= inv;
    const float4 c = xb[(long)l * 16 + d4];
    float4 se; se.x = c.x - sx; se.y = c.y - sy; se.z = c.z - sz; se.w = c.w - sw;
    ((float4*)(SEASON + bl * DM))[d4] = se;
    float4* ta = (float4*)(TACC + bl * DM);
    float4 t = ta[d4];
    t.x += sx; t.y += sy; t.z += sz; t.w += sw;
    ta[d4] = t;
}

// residual trend: rt[b,l] = sum_i sum_j TACC[b,(l+j-1)%L,i] * pw[i*3+j]
__global__ void circconv_kernel(const float* __restrict__ TACC, const float* __restrict__ pw,
                                float* __restrict__ RT)
{
    const long i = (long)blockIdx.x * 256 + threadIdx.x;   // TOK
    if (i >= TOK) return;
    const int l = (int)(i % LD), b = (int)(i / LD);
    float s = 0.f;
    for (int j = 0; j < 3; ++j) {
        int ll = l + j - 1; if (ll < 0) ll += LD; if (ll >= LD) ll -= LD;
        const float* row = TACC + ((long)b * LD + ll) * DM;
        for (int d = 0; d < DM; ++d) s += row[d] * pw[d * 3 + j];
    }
    RT[i] = s;
}

// per-token layernorm (part 1 of my_layernorm)
__global__ void ln_kernel(const float* __restrict__ X, const float* __restrict__ g,
                          const float* __restrict__ bb, float* __restrict__ XH)
{
    const long t = (long)blockIdx.x * 256 + threadIdx.x;
    if (t >= TOK) return;
    const float* x = X + t * DM;
    float s = 0.f, ss = 0.f;
    for (int d = 0; d < DM; ++d) { const float v = x[d]; s += v; ss += v * v; }
    const float mu = s * (1.0f / DM);
    const float var = ss * (1.0f / DM) - mu * mu;
    const float inv = rsqrtf(var + 1e-5f);
    float* y = XH + t * DM;
    for (int d = 0; d < DM; ++d) y[d] = (x[d] - mu) * inv * g[d] + bb[d];
}

// column mean over L per (b,d) (part 2 of my_layernorm)
__global__ void colmean_kernel(const float* __restrict__ XH, float* __restrict__ CM)
{
    const int i = blockIdx.x * 256 + threadIdx.x;          // BNB*DM
    if (i >= BNB * DM) return;
    const int d = i & 63, b = i >> 6;
    float s = 0.f;
    for (int l = 0; l < LD; ++l) s += XH[((long)b * LD + l) * DM + d];
    CM[i] = s * (1.0f / LD);
}

// out[b,lp] = sum_d (xh - colmean + trend)*pw[d] + rt*sum(pw) + pb
__global__ void final_kernel(const float* __restrict__ XH, const float* __restrict__ CM,
                             const float* __restrict__ TREND, const float* __restrict__ RT,
                             const float* __restrict__ pw, const float* __restrict__ pb,
                             float* __restrict__ out)
{
    const int i = blockIdx.x * 256 + threadIdx.x;          // BNB*PRED
    if (i >= BNB * PRED) return;
    const int lp = i % PRED, b = i / PRED;
    const int l = LD - PRED + lp;
    const float* xh = XH    + ((long)b * LD + l) * DM;
    const float* tr = TREND + ((long)b * LD + l) * DM;
    const float* cm = CM    + b * DM;
    float s = 0.f, sw = 0.f;
    for (int d = 0; d < DM; ++d) {
        s += (xh[d] - cm[d] + tr[d]) * pw[d];
        sw += pw[d];
    }
    out[i] = s + RT[(long)b * LD + l] * sw + pb[0];
}

// ---------------------------------------------------------------------------
extern "C" void kernel_launch(void* const* d_in, const int* in_sizes, int n_in,
                              void* d_out, int out_size, void* d_ws, size_t ws_size,
                              hipStream_t stream)
{
    const float* day   = (const float*)d_in[0];
    const float* c0w   = (const float*)d_in[1];  const float* c0b = (const float*)d_in[2];
    const float* c1w   = (const float*)d_in[3];  const float* c1b = (const float*)d_in[4];
    const float* c2w   = (const float*)d_in[5];  const float* c2b = (const float*)d_in[6];
    const float* sawq  = (const float*)d_in[7];  const float* sabq = (const float*)d_in[8];
    const float* sawk  = (const float*)d_in[9];  const float* sabk = (const float*)d_in[10];
    const float* sawv  = (const float*)d_in[11]; const float* sabv = (const float*)d_in[12];
    const float* sawo  = (const float*)d_in[13]; const float* sabo = (const float*)d_in[14];
    const float* cawq  = (const float*)d_in[15]; const float* cabq = (const float*)d_in[16];
    const float* cawk  = (const float*)d_in[17]; const float* cabk = (const float*)d_in[18];
    const float* cawv  = (const float*)d_in[19]; const float* cabv = (const float*)d_in[20];
    const float* cawo  = (const float*)d_in[21]; const float* cabo = (const float*)d_in[22];
    const float* ff1w  = (const float*)d_in[23]; const float* ff2w = (const float*)d_in[24];
    const float* projw = (const float*)d_in[25];
    const float* lng   = (const float*)d_in[26]; const float* lnb  = (const float*)d_in[27];
    const float* predw = (const float*)d_in[28]; const float* predb= (const float*)d_in[29];
    float* out = (float*)d_out;

    const long BIG = (long)TOK * DM;               // 14,155,776 floats
    float* B0 = (float*)d_ws;                      // xd (s1) / ping-pong
    float* B1 = B0 + BIG;                          // cross (s2) / late season
    float* B2 = B1 + BIG;                          // trend (s3)
    float* B3 = B2 + BIG;                          // TACC = t1+t2+t3
    float* B4 = B3 + BIG;                          // q / season / xd+y
    float* B5 = B4 + BIG;                          // k / FFN hidden (B5+B6)
    float* B6 = B5 + BIG;
    float* B7 = B6 + BIG;                          // v / agg
    float* SM = B7 + BIG;                          // small scratch region
    float* MV = SM;                                // [BNB*LD]
    float* CW = MV + (long)BNB * LD;               // [BNB*4]
    float* RT = CW + BNB * NTOPK;                  // [BNB*LD]
    float* CM = RT + (long)BNB * LD;               // [BNB*DM]
    int*  IDX = (int*)(CM + BNB * DM);             // [4]

    const int BLK = 256;
    const int gBig   = (int)(((long)TOK * DM) / BLK);        // 55296
    const int gVec   = (int)(((long)TOK * 16 + BLK - 1) / BLK); // 13824
    const int gTok   = (TOK + BLK - 1) / BLK;                // 864
    const int gCM    = (BNB * DM + BLK - 1) / BLK;           // 128
    const int gOut   = (BNB * PRED + BLK - 1) / BLK;         // 288
    const int gT64   = (int)(((long)(TOK / 16) * (64 / 16)  + 7) / 8);  // wmma N=64
    const int gT128  = (int)(((long)(TOK / 16) * (128 / 16) + 7) / 8);  // wmma N=128
    const size_t CORR_SHM = (size_t)(2 * LD * CPAD + LD) * sizeof(float); // 236,736B

    // conv stems
    conv0_kernel<<<gBig, BLK, 0, stream>>>(day, c0w, c0b, B0);
    conv1_kernel<<<gBig, BLK, 0, stream>>>(day, c1w, c1b, B1);
    conv2_kernel<<<gBig, BLK, 0, stream>>>(day, c2w, c2b, B2);
    hipMemsetAsync(B3, 0, BIG * sizeof(float), stream);      // TACC = 0

    // ---- self attention (auto-correlation) ----
    linear_wmma<64, 64, true,  false, false><<<gT64, BLK, 0, stream>>>(B0, sawq, sabq, nullptr, B4, TOK);
    linear_wmma<64, 64, true,  false, false><<<gT64, BLK, 0, stream>>>(B0, sawk, sabk, nullptr, B5, TOK);
    linear_wmma<64, 64, true,  false, false><<<gT64, BLK, 0, stream>>>(B0, sawv, sabv, nullptr, B6, TOK);
    corr_wmma_kernel<<<BNB, 1024, CORR_SHM, stream>>>(B4, B5, MV);
    topk_kernel<<<1, 512, 0, stream>>>(MV, IDX);
    softmax_kernel<<<(BNB + BLK - 1) / BLK, BLK, 0, stream>>>(MV, IDX, CW);
    agg_kernel<<<gVec, BLK, 0, stream>>>(B6, IDX, CW, B7);
    linear_wmma<64, 64, true,  false, true ><<<gT64, BLK, 0, stream>>>(B7, sawo, sabo, B0, B0, TOK);
    decomp_kernel<<<gVec, BLK, 0, stream>>>(B0, B4, B3);     // xd -> B4, t1 -> TACC

    // ---- cross attention ----
    linear_wmma<64, 64, true,  false, false><<<gT64, BLK, 0, stream>>>(B4, cawq, cabq, nullptr, B5, TOK);
    linear_wmma<64, 64, true,  false, false><<<gT64, BLK, 0, stream>>>(B1, cawk, cabk, nullptr, B6, TOK);
    linear_wmma<64, 64, true,  false, false><<<gT64, BLK, 0, stream>>>(B1, cawv, cabv, nullptr, B7, TOK);
    corr_wmma_kernel<<<BNB, 1024, CORR_SHM, stream>>>(B5, B6, MV);
    topk_kernel<<<1, 512, 0, stream>>>(MV, IDX);
    softmax_kernel<<<(BNB + BLK - 1) / BLK, BLK, 0, stream>>>(MV, IDX, CW);
    agg_kernel<<<gVec, BLK, 0, stream>>>(B7, IDX, CW, B0);
    linear_wmma<64, 64, true,  false, true ><<<gT64, BLK, 0, stream>>>(B0, cawo, cabo, B4, B4, TOK);
    decomp_kernel<<<gVec, BLK, 0, stream>>>(B4, B0, B3);     // xd -> B0, t2 -> TACC

    // ---- FFN (hidden aliases freed B5+B6 = exactly TOK*128 floats) ----
    float* H = B5;
    linear_wmma<64, 128, false, true,  false><<<gT128, BLK, 0, stream>>>(B0, ff1w, nullptr, nullptr, H, TOK);
    linear_wmma<128, 64, false, false, true ><<<gT64,  BLK, 0, stream>>>(H, ff2w, nullptr, B0, B4, TOK);
    decomp_kernel<<<gVec, BLK, 0, stream>>>(B4, B1, B3);     // xd -> B1, t3 -> TACC

    // ---- trend path + layernorm + head ----
    circconv_kernel<<<gTok, BLK, 0, stream>>>(B3, projw, RT);
    ln_kernel<<<gTok, BLK, 0, stream>>>(B1, lng, lnb, B4);
    colmean_kernel<<<gCM, BLK, 0, stream>>>(B4, CM);
    final_kernel<<<gOut, BLK, 0, stream>>>(B4, CM, B2, RT, predw, predb, out);
}
